// BandMergeDecoder_68848325755444
// MI455X (gfx1250) — matrix operand order repaired and checked by reading, not compiled, not run on hardware
//
#include <hip/hip_runtime.h>
#include <hip/hip_bf16.h>

typedef _Float16 half_t;
typedef __attribute__((ext_vector_type(16))) _Float16 v16h;
typedef __attribute__((ext_vector_type(8)))  _Float16 v8h;
typedef __attribute__((ext_vector_type(8)))  float    v8f;
typedef __attribute__((ext_vector_type(4)))  float    v4f;

#define NBANDS 22
#define DDIM   128
#define TFRAMES 2048
#define BATCH   8
#define MROWS  (BATCH * TFRAMES)     // 16384
#define FREQ   1025
#define KPAD   2080                  // 2*1025 padded to %32
#define NFFT   2048
#define HOP    512
#define OUTLEN (HOP * (TFRAMES - 1)) // 1048064 per batch

#define TWO_PI_F 6.28318530717958647692f

static __device__ __forceinline__ v8f wmma_f16(v16h a, v16h b, v8f c) {
  // D = A(16x32 f16) * B(32x16 f16) + C(16x16 f32)
  return __builtin_amdgcn_wmma_f32_16x16x32_f16(
      false, a, false, b, (short)0, c, false, false);
}

// ---------------------------------------------------------------------------
// Kernel 0: windowed irfft basis, f16 [KPAD x NFFT].
// row 2k   -> (w_k/N) * cos(2*pi*k*n/N) * hann(n)
// row 2k+1 -> -(w_k/N) * sin(2*pi*k*n/N) * hann(n)   (0 for k==0, k==N/2)
// rows >= 2050 -> 0 (K padding)
// ---------------------------------------------------------------------------
__global__ void gen_basis_kernel(half_t* __restrict__ basis) {
  int idx = blockIdx.x * 256 + threadIdx.x;   // KPAD*NFFT total
  if (idx >= KPAD * NFFT) return;
  int nn = idx & (NFFT - 1);
  int k2 = idx >> 11;
  int kk = k2 >> 1;
  int im = k2 & 1;
  float val = 0.0f;
  if (kk <= NFFT / 2) {
    float w = (kk == 0 || kk == NFFT / 2) ? (1.0f / NFFT) : (2.0f / NFFT);
    int m = (kk * nn) & (NFFT - 1);           // exact angle reduction
    float ang = (float)m * (TWO_PI_F / NFFT);
    if (!im)
      val = w * cosf(ang);
    else if (kk != 0 && kk != NFFT / 2)
      val = -w * sinf(ang);
    float win = 0.5f * (1.0f - cosf((float)nn * (TWO_PI_F / NFFT)));
    val *= win;
  }
  basis[idx] = (half_t)val;
}

// ---------------------------------------------------------------------------
// Kernel 1: per-band GEMM  band[m, n, :] = z[m, n, :] @ W[n] + bias[n]
// WMMA f16 fragments converted on the fly from f32 inputs; f32 accumulate.
// grid = (MROWS/128, NBANDS), block = 256 (8 waves). Wave w owns a 16-row
// M sub-tile and sweeps all 8 N tiles; A fragments preloaded for 4 K-steps.
// ---------------------------------------------------------------------------
__global__ void band_gemm_kernel(const float* __restrict__ z,
                                 const float* __restrict__ W,
                                 const float* __restrict__ bias,
                                 half_t* __restrict__ bandbuf) {
  int lane = threadIdx.x & 31;
  int wave = threadIdx.x >> 5;
  int n    = blockIdx.y;
  int m0   = blockIdx.x * 128 + wave * 16;

  int rowA = m0 + (lane & 15);
  const float* zrow = z + ((size_t)rowA * NBANDS + n) * DDIM;
  int khalf = (lane >> 4) * 8;

  v16h afr[4];
#pragma unroll
  for (int ks = 0; ks < 4; ++ks) {
    int kb = ks * 32 + khalf;
    float tmp[16] __attribute__((aligned(16)));
    *(v4f*)(tmp + 0)  = *(const v4f*)(zrow + kb + 0);
    *(v4f*)(tmp + 4)  = *(const v4f*)(zrow + kb + 4);
    *(v4f*)(tmp + 8)  = *(const v4f*)(zrow + kb + 16);
    *(v4f*)(tmp + 12) = *(const v4f*)(zrow + kb + 20);
    v16h a;
#pragma unroll
    for (int h = 0; h < 16; ++h) a[h] = (half_t)tmp[h];
    afr[ks] = a;
  }

  int mrow = m0 + (lane >> 4) * 8;
#pragma unroll
  for (int j = 0; j < 8; ++j) {
    v8f acc = {};
#pragma unroll
    for (int ks = 0; ks < 4; ++ks) {
      int k = ks * 32 + (lane >> 4) * 8;         // B col = lane%16, k in table order
      const float* wr = W + ((size_t)n * DDIM + k) * DDIM + j * 16 + (lane & 15);
      float tmp[16] __attribute__((aligned(16)));
#pragma unroll
      for (int s = 0; s < 16; ++s) tmp[s] = wr[(size_t)(s < 8 ? s : s + 8) * DDIM];
      v16h bfr;
#pragma unroll
      for (int h = 0; h < 16; ++h) bfr[h] = (half_t)tmp[h];
      acc = wmma_f16(afr[ks], bfr, acc);
    }
    int col = j * 16 + (lane & 15);
    float bv = bias[n * DDIM + col];
#pragma unroll
    for (int r = 0; r < 8; ++r) {
      bandbuf[((size_t)(mrow + r) * NBANDS + n) * DDIM + col] =
          (half_t)(acc[r] + bv);
    }
  }
}

// ---------------------------------------------------------------------------
// Kernel 2: gather overlapping bands -> mask, normalize by coverage count,
// complex multiply with mix_spec, pack f16 S[MROWS x KPAD] (zero K-pad).
// Band n in [0,20] covers freqs [48n, 48n+63]; band 21 covers [961,1024].
// grid = MROWS blocks of 256 threads.
// ---------------------------------------------------------------------------
__global__ void mask_mix_kernel(const half_t* __restrict__ bandbuf,
                                const float* __restrict__ mix,
                                half_t* __restrict__ S) {
  int row = blockIdx.x;
  int b = row >> 11;
  int t = row & (TFRAMES - 1);
  for (int f = threadIdx.x; f < KPAD / 2; f += 256) {
    size_t so = (size_t)row * KPAD + 2 * f;
    if (f < FREQ) {
      int nhi = f / 48;        if (nhi > 20) nhi = 20;
      int nlo = (f - 16) / 48; if (nlo < 0)  nlo = 0;
      float mr = 0.0f, mi = 0.0f;
      int cnt = 0;
      for (int nb = nlo; nb <= nhi; ++nb) {
        int col = f - 48 * nb;
        const half_t* p = bandbuf + ((size_t)row * NBANDS + nb) * DDIM + 2 * col;
        mr += (float)p[0]; mi += (float)p[1]; ++cnt;
      }
      if (f >= 961) {
        int col = f - 961;
        const half_t* p = bandbuf + ((size_t)row * NBANDS + 21) * DDIM + 2 * col;
        mr += (float)p[0]; mi += (float)p[1]; ++cnt;
      }
      float inv = 1.0f / (float)cnt;
      mr *= inv; mi *= inv;
      float xr = mix[(((size_t)b * 2 + 0) * TFRAMES + t) * FREQ + f];
      float xi = mix[(((size_t)b * 2 + 1) * TFRAMES + t) * FREQ + f];
      S[so + 0] = (half_t)(mr * xr - mi * xi);
      S[so + 1] = (half_t)(mr * xi + mi * xr);
    } else {
      S[so + 0] = (half_t)0.0f;
      S[so + 1] = (half_t)0.0f;
    }
  }
}

// ---------------------------------------------------------------------------
// Kernel 3: ISTFT GEMM  frames[M x NFFT] = S[M x KPAD] @ basis[KPAD x NFFT]
// (window folded into basis).
// Block tile 128 rows x 256 cols, 8 waves in a 2x4 grid; each wave computes
// 4 M-tiles x 4 N-tiles = 16 accumulators. Per K-step: 8 A loads + 64 B element loads
// feed 16 WMMAs (4.5 VMEM per WMMA). Basis (8.5 MB) is L2-resident; S is
// streamed once with explicit prefetch of the next K-step.
// grid = (NFFT/256, MROWS/128), block = 256.
// ---------------------------------------------------------------------------
__global__ void istft_gemm_kernel(const half_t* __restrict__ S,
                                  const half_t* __restrict__ basis,
                                  half_t* __restrict__ frames) {
  int lane = threadIdx.x & 31;
  int wave = threadIdx.x >> 5;
  int m0  = blockIdx.y * 128 + (wave & 1) * 64;   // 4 M-tiles: m0 .. m0+63
  int nb0 = blockIdx.x * 256 + (wave >> 1) * 64;  // 4 N-tiles: nb0 .. nb0+63

  v8f acc[16] = {};

  // Per-lane A base pointers for the 4 M-tiles
  const half_t* srow[4];
#pragma unroll
  for (int mi = 0; mi < 4; ++mi) {
    srow[mi] = S + (size_t)(m0 + mi * 16 + (lane & 15)) * KPAD + ((lane >> 4) * 8);
  }

  for (int ks = 0; ks < KPAD / 32; ++ks) {
    int kb = ks * 32;

    v16h a[4];
#pragma unroll
    for (int mi = 0; mi < 4; ++mi) {
      v8h alo = *(const v8h*)(srow[mi] + kb);
      v8h ahi = *(const v8h*)(srow[mi] + kb + 16);
      a[mi] = __builtin_shufflevector(alo, ahi,
                                      0, 1, 2, 3, 4, 5, 6, 7,
                                      8, 9, 10, 11, 12, 13, 14, 15);
      // prefetch next K-step of the streaming operand (global_prefetch_b8)
      __builtin_prefetch(srow[mi] + kb + 32, 0, 3);
    }

    const half_t* brow = basis + (size_t)(kb + (lane >> 4) * 8) * NFFT + nb0 + (lane & 15);
    v16h bfr[4];
#pragma unroll
    for (int ni = 0; ni < 4; ++ni) {
#pragma unroll
      for (int s = 0; s < 16; ++s)
        bfr[ni][s] = brow[(size_t)(s < 8 ? s : s + 8) * NFFT + ni * 16];
    }

#pragma unroll
    for (int mi = 0; mi < 4; ++mi) {
#pragma unroll
      for (int ni = 0; ni < 4; ++ni) {
        acc[mi * 4 + ni] = wmma_f16(a[mi], bfr[ni], acc[mi * 4 + ni]);
      }
    }
  }

#pragma unroll
  for (int mi = 0; mi < 4; ++mi) {
    int mrow = m0 + mi * 16 + (lane >> 4) * 8;
#pragma unroll
    for (int ni = 0; ni < 4; ++ni) {
      int colb = nb0 + ni * 16 + (lane & 15);
#pragma unroll
      for (int r = 0; r < 8; ++r) {
        frames[(size_t)(mrow + r) * NFFT + colb] = (half_t)acc[mi * 4 + ni][r];
      }
    }
  }
}

// ---------------------------------------------------------------------------
// Kernel 4: overlap-add gather + exact Hann^2 envelope + crop(1024).
// Each output sample has <=4 contributing frames; no atomics (deterministic).
// ---------------------------------------------------------------------------
__global__ void ola_kernel(const half_t* __restrict__ frames,
                           float* __restrict__ out) {
  size_t tid = (size_t)blockIdx.x * 256 + threadIdx.x;
  if (tid >= (size_t)BATCH * OUTLEN) return;
  int b = (int)(tid / OUTLEN);
  int i = (int)(tid % OUTLEN);
  int p = i + NFFT / 2;
  int jmax = p >> 9; if (jmax > TFRAMES - 1) jmax = TFRAMES - 1;
  int jmin = jmax - 3; if (jmin < 0) jmin = 0;
  float sum = 0.0f, env = 0.0f;
  for (int j = jmin; j <= jmax; ++j) {
    int nn = p - (j << 9);
    float w = 0.5f * (1.0f - cosf((float)nn * (TWO_PI_F / NFFT)));
    env += w * w;
    sum += (float)frames[((size_t)(b * TFRAMES + j)) * NFFT + nn];
  }
  out[tid] = sum / (env > 1e-11f ? env : 1.0f);
}

// ---------------------------------------------------------------------------
// Launcher. Workspace layout (bytes):
//   [0, 8519680)              basis f16 [2080 x 2048]
//   [8519680, 76677120)       S     f16 [16384 x 2080]
//   [76677120, ...)           region reused: bandbuf f16 [16384x22x128]
//                             then frames f16 [16384 x 2048] (bandbuf dead)
// Total ~169 MB.
// ---------------------------------------------------------------------------
extern "C" void kernel_launch(void* const* d_in, const int* in_sizes, int n_in,
                              void* d_out, int out_size, void* d_ws, size_t ws_size,
                              hipStream_t stream) {
  const float* z    = (const float*)d_in[0];  // [8,2048,22,128]
  const float* mix  = (const float*)d_in[1];  // [8,2,2048,1025]
  const float* W    = (const float*)d_in[2];  // [22,128,128]
  const float* bias = (const float*)d_in[3];  // [22,128]
  float* out = (float*)d_out;                 // [8,1048064]

  char* ws = (char*)d_ws;
  half_t* basis   = (half_t*)(ws);
  half_t* S       = (half_t*)(ws + (size_t)8519680);
  half_t* region2 = (half_t*)(ws + (size_t)76677120);
  half_t* bandbuf = region2;
  half_t* frames  = region2;   // reuses bandbuf storage after it is consumed

  (void)in_sizes; (void)n_in; (void)out_size; (void)ws_size;

  gen_basis_kernel<<<(KPAD * NFFT) / 256, 256, 0, stream>>>(basis);
  band_gemm_kernel<<<dim3(MROWS / 128, NBANDS), 256, 0, stream>>>(z, W, bias, bandbuf);
  mask_mix_kernel<<<MROWS, 256, 0, stream>>>(bandbuf, mix, S);
  istft_gemm_kernel<<<dim3(NFFT / 256, MROWS / 128), 256, 0, stream>>>(S, basis, frames);
  ola_kernel<<<(unsigned)(((size_t)BATCH * OUTLEN + 255) / 256), 256, 0, stream>>>(frames, out);
}
